// D_CLEM_60473139528288
// MI455X (gfx1250) — compile-verified, matrix operand order checked by reading, not run on hardware
//
#include <hip/hip_runtime.h>
#include <math.h>

// ---------------- CDNA5 WMMA plumbing (wave32) ----------------
typedef __attribute__((ext_vector_type(16))) __bf16 v16bf;
typedef __attribute__((ext_vector_type(8)))  float  v8f;

union FragAB { v16bf v; unsigned short u[16]; uint4 q[2]; };

__device__ __forceinline__ unsigned short f2bf(float f) {
  union { float f; unsigned u; } c; c.f = f;
  unsigned r = 0x7FFFu + ((c.u >> 16) & 1u);          // RNE
  return (unsigned short)((c.u + r) >> 16);
}
__device__ __forceinline__ float bf2f(unsigned short u) {
  union { unsigned u; float f; } c; c.u = ((unsigned)u) << 16;
  return c.f;
}

// element j of a lane's A/B fragment holds K index (16-bit 16x32 layout,
// CDNA5 ISA 7.12.2):  j<8 -> j+8*half ; j>=8 -> 16+(j-8)+8*half.
// => u[0..7] and u[8..15] are two contiguous 8-element (16B) runs of K.
__device__ __forceinline__ void ld_frag(FragAB& f, const unsigned short* p) {
  f.q[0] = *(const uint4*)(p);
  f.q[1] = *(const uint4*)(p + 16);
}
__device__ __forceinline__ void zero_frag(FragAB& f) {
  f.q[0] = make_uint4(0, 0, 0, 0);
  f.q[1] = make_uint4(0, 0, 0, 0);
}

__device__ __forceinline__ float silu_f(float v) {
  return v / (1.0f + __expf(-v));
}

// ---------------- problem constants ----------------
constexpr int B_  = 4;
constexpr int C_  = 256;
constexpr int H_  = 64;
constexpr int W_  = 64;
constexpr int HW  = H_ * W_;          // 4096
constexpr int BHW = B_ * HW;          // 16384
constexpr int NT  = BHW / 16;         // 1024 spatial tiles

// =====================================================================
// prep: conversions / layout shuffles (bandwidth-trivial: ~60 MB total)
// =====================================================================
__global__ void k_cvt_bf16(const float* __restrict__ in,
                           unsigned short* __restrict__ out, int n) {
  int i = blockIdx.x * blockDim.x + threadIdx.x;
  if (i < n) out[i] = f2bf(in[i]);
}

// NCHW f32 -> NHWC bf16  (out[((b*HW+p)*C)+c])
__global__ void k_nchw2nhwc(const float* __restrict__ in,
                            unsigned short* __restrict__ out, int C, int n) {
  int o = blockIdx.x * blockDim.x + threadIdx.x;
  if (o >= n) return;
  int c = o % C;
  int t = o / C;
  int p = t % HW;
  int b = t / HW;
  out[o] = f2bf(in[((size_t)b * C + c) * HW + p]);
}

// w[(co*CIN+ci)*9+tap] -> out[(co*9+tap)*CIN+ci], rows co>=COUT zeroed (pad)
__global__ void k_w3x3_reorder(const float* __restrict__ w,
                               unsigned short* __restrict__ out,
                               int COUT, int CIN, int n) {
  int o = blockIdx.x * blockDim.x + threadIdx.x;
  if (o >= n) return;
  int ci  = o % CIN;
  int t   = o / CIN;
  int tap = t % 9;
  int co  = t / 9;
  out[o] = (co < COUT) ? f2bf(w[((size_t)co * CIN + ci) * 9 + tap]) : (unsigned short)0;
}

// =====================================================================
// 3x3 conv, pad=1, Cin=256, tap-major implicit im2col (K = tap*256+ci).
// One wave per spatial tile carries all MT co-tiles; B fragment built
// once per 32-K chunk (2 x b128, tap uniform per chunk) and reused.
// BN=false: f32 NCHW store (offset head). BN=true: BN+SiLU, bf16 NHWC.
// =====================================================================
template<int COUT, int MT, bool BN>
__global__ void k_conv3x3(const unsigned short* __restrict__ in,   // NHWC bf16
                          const unsigned short* __restrict__ w,    // (pad,9,256) bf16
                          const float* __restrict__ bias,
                          const float* __restrict__ gamma,
                          const float* __restrict__ beta,
                          const float* __restrict__ mean,
                          const float* __restrict__ var,
                          float* __restrict__ outf,                // BN=false
                          unsigned short* __restrict__ outbf)      // BN=true
{
  const int gw   = (blockIdx.x * blockDim.x + threadIdx.x) >> 5;
  const int lane = threadIdx.x & 31;
  if (gw >= NT) return;

  const int p0  = gw * 16;
  const int b   = p0 / HW;
  const int rem = p0 % HW;
  const int y   = rem / W_;
  const int x0  = rem % W_;
  const int half = lane >> 4;
  const int ln   = lane & 15;
  const int px   = x0 + ln;

  v8f acc[MT] = {};
  for (int kb = 0; kb < 9 * 256; kb += 32) {
    const int tap = kb >> 8;          // uniform per chunk
    const int cb  = kb & 255;
    const int ky  = tap / 3 - 1;
    const int kx  = tap - (tap / 3) * 3 - 1;
    const int yy  = y + ky;
    if (yy < 0 || yy >= H_) continue;            // wave-uniform skip
    const int xx = px + kx;
    FragAB fb;
    if (xx >= 0 && xx < W_) {
      ld_frag(fb, in + ((size_t)(b * HW + yy * W_ + xx) * 256 + cb + 8 * half));
    } else {
      zero_frag(fb);
    }
#pragma unroll
    for (int ct = 0; ct < MT; ++ct) {
      FragAB fa;
      ld_frag(fa, w + ((size_t)(ct * 16 + ln) * 2304 + kb + 8 * half));
      acc[ct] = __builtin_amdgcn_wmma_f32_16x16x32_bf16(
          false, fa.v, false, fb.v, (short)0, acc[ct], false, false);
    }
  }

#pragma unroll
  for (int ct = 0; ct < MT; ++ct) {
#pragma unroll
    for (int r = 0; r < 8; ++r) {
      const int co = ct * 16 + r + half * 8;
      if (co < COUT) {
        float v = acc[ct][r] + bias[co];
        if (BN) {
          const float inv = gamma[co] * __frsqrt_rn(var[co] + 1e-5f);
          v = silu_f(v * inv + (beta[co] - mean[co] * inv));
          outbf[(size_t)(b * HW + y * W_ + x0 + ln) * COUT + co] = f2bf(v);
        } else {
          outf[((size_t)b * COUT + co) * HW + y * W_ + x0 + ln] = v;
        }
      }
    }
  }
}

// =====================================================================
// Deformable grouped conv. One wave per (spatial tile, group); 4 co-tile
// accumulators share each gathered B fragment. K = tap*64 + ci (tap
// uniform per 32-chunk -> bilinear state fetched 4x/chunk from LDS,
// corners are conditional b128 loads of 8 channels).
// =====================================================================
__global__ void k_deform(const unsigned short* __restrict__ xbf,  // NHWC bf16
                         const float* __restrict__ off,           // (B,18,H,W) f32
                         const unsigned short* __restrict__ wdef, // (256,9,64) bf16
                         unsigned short* __restrict__ xdir)       // NHWC bf16
{
  __shared__ int   tyi[128 * 9];
  __shared__ int   txi[128 * 9];
  __shared__ float twy[128 * 9];
  __shared__ float twx[128 * 9];

  const int gw   = (blockIdx.x * blockDim.x + threadIdx.x) >> 5;
  const int lane = threadIdx.x & 31;
  const int tid9 = (int)threadIdx.x * 9;
  if (gw >= NT * 4) return;

  const int st = gw >> 2;
  const int g  = gw & 3;
  const int p0  = st * 16;
  const int b   = p0 / HW;
  const int rem = p0 % HW;
  const int y   = rem / W_;
  const int x0  = rem % W_;
  const int half = lane >> 4;
  const int ln   = lane & 15;
  const int px   = x0 + ln;

#pragma unroll
  for (int t = 0; t < 9; ++t) {
    const float dy = off[(((size_t)b * 18 + 2 * t + 0) * HW) + y * W_ + px];
    const float dx = off[(((size_t)b * 18 + 2 * t + 1) * HW) + y * W_ + px];
    const float py = (float)y  + (float)(t / 3 - 1) + dy;
    const float qx = (float)px + (float)(t % 3 - 1) + dx;
    const float fy = floorf(py), fx = floorf(qx);
    tyi[tid9 + t] = (int)fy;
    txi[tid9 + t] = (int)fx;
    twy[tid9 + t] = py - fy;
    twx[tid9 + t] = qx - fx;
  }

  v8f acc[4] = {};
  for (int kb = 0; kb < 576; kb += 32) {
    const int tap = kb >> 6;          // uniform per chunk
    const int cb  = kb & 63;
    const int y0  = tyi[tid9 + tap];
    const int xx0 = txi[tid9 + tap];
    const float wy = twy[tid9 + tap];
    const float wx = twx[tid9 + tap];
    const float w00 = (1.0f - wy) * (1.0f - wx);
    const float w01 = (1.0f - wy) * wx;
    const float w10 = wy * (1.0f - wx);
    const float w11 = wy * wx;
    const bool yv0 = ((unsigned)y0 < (unsigned)H_);
    const bool yv1 = ((unsigned)(y0 + 1) < (unsigned)H_);
    const bool xv0 = ((unsigned)xx0 < (unsigned)W_);
    const bool xv1 = ((unsigned)(xx0 + 1) < (unsigned)W_);

    FragAB c00, c01, c10, c11;
    const int choff = g * 64 + cb + 8 * half;
    auto ldc = [&](int yy, int xs, bool ok, FragAB& f) {
      if (ok) ld_frag(f, xbf + ((size_t)(b * HW + yy * W_ + xs) * 256 + choff));
      else    zero_frag(f);
    };
    ldc(y0,     xx0,     yv0 && xv0, c00);
    ldc(y0,     xx0 + 1, yv0 && xv1, c01);
    ldc(y0 + 1, xx0,     yv1 && xv0, c10);
    ldc(y0 + 1, xx0 + 1, yv1 && xv1, c11);

    FragAB fb;
#pragma unroll
    for (int e = 0; e < 16; ++e) {
      float v = bf2f(c00.u[e]) * w00;
      v = fmaf(bf2f(c01.u[e]), w01, v);
      v = fmaf(bf2f(c10.u[e]), w10, v);
      v = fmaf(bf2f(c11.u[e]), w11, v);
      fb.u[e] = f2bf(v);
    }
#pragma unroll
    for (int ct = 0; ct < 4; ++ct) {
      FragAB fa;
      ld_frag(fa, wdef + ((size_t)(g * 64 + ct * 16 + ln) * 576 + kb + 8 * half));
      acc[ct] = __builtin_amdgcn_wmma_f32_16x16x32_bf16(
          false, fa.v, false, fb.v, (short)0, acc[ct], false, false);
    }
  }

#pragma unroll
  for (int ct = 0; ct < 4; ++ct) {
#pragma unroll
    for (int r = 0; r < 8; ++r) {
      const int co = g * 64 + ct * 16 + r + half * 8;
      xdir[(size_t)(b * HW + y * W_ + x0 + ln) * 256 + co] = f2bf(acc[ct][r]);
    }
  }
}

// =====================================================================
// x_dense = w_cross(256x512) @ concat(x_dir, x_prev). One wave per
// (spatial tile, 64-row M-block): 4 co-tiles share each B fragment.
// Stores x_dense both f32 (for the gate) and bf16 (for g1) in NHWC.
// =====================================================================
__global__ void k_dense(const unsigned short* __restrict__ xdir,   // NHWC bf16
                        const unsigned short* __restrict__ xprev,  // NHWC bf16
                        const unsigned short* __restrict__ wc,     // (256,512) bf16
                        float* __restrict__ xdense_f,              // NHWC f32
                        unsigned short* __restrict__ xdense_bf)    // NHWC bf16
{
  const int gw   = (blockIdx.x * blockDim.x + threadIdx.x) >> 5;
  const int lane = threadIdx.x & 31;
  if (gw >= NT * 4) return;

  const int st = gw >> 2;
  const int mb = gw & 3;                 // 64-row M block
  const int p0  = st * 16;
  const int b   = p0 / HW;
  const int rem = p0 % HW;
  const int y   = rem / W_;
  const int x0  = rem % W_;
  const int half = lane >> 4;
  const int ln   = lane & 15;
  const size_t pos = (size_t)(b * HW + y * W_ + x0 + ln);

  v8f acc[4] = {};
  for (int kb = 0; kb < 512; kb += 32) {
    FragAB fb;
    const unsigned short* src = (kb < 256)
        ? (xdir  + pos * 256 + kb + 8 * half)
        : (xprev + pos * 256 + (kb - 256) + 8 * half);
    ld_frag(fb, src);
#pragma unroll
    for (int ct = 0; ct < 4; ++ct) {
      FragAB fa;
      ld_frag(fa, wc + ((size_t)(mb * 64 + ct * 16 + ln) * 512 + kb + 8 * half));
      acc[ct] = __builtin_amdgcn_wmma_f32_16x16x32_bf16(
          false, fa.v, false, fb.v, (short)0, acc[ct], false, false);
    }
  }
#pragma unroll
  for (int ct = 0; ct < 4; ++ct) {
#pragma unroll
    for (int r = 0; r < 8; ++r) {
      const int co = mb * 64 + ct * 16 + r + half * 8;
      const float v = acc[ct][r];
      xdense_f [pos * 256 + co] = v;
      xdense_bf[pos * 256 + co] = f2bf(v);
    }
  }
}

// =====================================================================
// Fused tail: attn = sigmoid(w_g2 @ a), z = x_dense*attn staged in LDS
// ([n][k] + pad so phase-2 B = ds_load_b128), out = silu(bn(w_out@z))+x.
// One block (8 waves) per spatial tile; each wave: 2 co-tiles/phase.
// =====================================================================
__global__ void k_attn_out(const unsigned short* __restrict__ abuf, // NHWC(64) bf16
                           const unsigned short* __restrict__ wg2,  // (256,64) bf16
                           const float* __restrict__ bg2,
                           const float* __restrict__ xdense_f,      // NHWC f32
                           const unsigned short* __restrict__ wout, // (256,256) bf16
                           const float* __restrict__ bout,
                           const float* __restrict__ og,
                           const float* __restrict__ ob,
                           const float* __restrict__ om,
                           const float* __restrict__ ov,
                           const float* __restrict__ xin,           // NCHW f32
                           float* __restrict__ outp)                // NCHW f32
{
  __shared__ unsigned short zlds[16 * 264];   // [n][k], 264 = 256 + 8 pad

  const int st   = blockIdx.x;
  const int wid  = threadIdx.x >> 5;
  const int lane = threadIdx.x & 31;
  const int half = lane >> 4;
  const int ln   = lane & 15;

  const int p0  = st * 16;
  const int b   = p0 / HW;
  const int rem = p0 % HW;
  const int y   = rem / W_;
  const int x0  = rem % W_;
  const size_t pos = (size_t)(b * HW + y * W_ + x0 + ln);

  // ---- phase 1: attn (K=64) + gate ----
  {
    v8f acc[2] = {};
    for (int kb = 0; kb < 64; kb += 32) {
      FragAB fb;
      ld_frag(fb, abuf + pos * 64 + kb + 8 * half);
#pragma unroll
      for (int t = 0; t < 2; ++t) {
        FragAB fa;
        ld_frag(fa, wg2 + ((size_t)((wid * 2 + t) * 16 + ln) * 64 + kb + 8 * half));
        acc[t] = __builtin_amdgcn_wmma_f32_16x16x32_bf16(
            false, fa.v, false, fb.v, (short)0, acc[t], false, false);
      }
    }
#pragma unroll
    for (int t = 0; t < 2; ++t) {
#pragma unroll
      for (int r = 0; r < 8; ++r) {
        const int co = (wid * 2 + t) * 16 + r + half * 8;
        const float attn = 1.0f / (1.0f + __expf(-(acc[t][r] + bg2[co])));
        const float z = xdense_f[pos * 256 + co] * attn;
        zlds[ln * 264 + co] = f2bf(z);
      }
    }
  }
  __syncthreads();

  // ---- phase 2: out = silu(bn(w_out @ z)) + x, K=256 from LDS ----
  {
    v8f acc[2] = {};
    for (int kb = 0; kb < 256; kb += 32) {
      FragAB fb;
      const unsigned short* zp = &zlds[ln * 264 + kb + 8 * half];
      fb.q[0] = *(const uint4*)(zp);
      fb.q[1] = *(const uint4*)(zp + 16);
#pragma unroll
      for (int t = 0; t < 2; ++t) {
        FragAB fa;
        ld_frag(fa, wout + ((size_t)((wid * 2 + t) * 16 + ln) * 256 + kb + 8 * half));
        acc[t] = __builtin_amdgcn_wmma_f32_16x16x32_bf16(
            false, fa.v, false, fb.v, (short)0, acc[t], false, false);
      }
    }
#pragma unroll
    for (int t = 0; t < 2; ++t) {
#pragma unroll
      for (int r = 0; r < 8; ++r) {
        const int co = (wid * 2 + t) * 16 + r + half * 8;
        float v = acc[t][r] + bout[co];
        const float inv = og[co] * __frsqrt_rn(ov[co] + 1e-5f);
        v = silu_f(v * inv + (ob[co] - om[co] * inv));
        const size_t idx = ((size_t)b * 256 + co) * HW + y * W_ + x0 + ln;
        outp[idx] = xin[idx] + v;
      }
    }
  }
}

// =====================================================================
extern "C" void kernel_launch(void* const* d_in, const int* in_sizes, int n_in,
                              void* d_out, int out_size, void* d_ws, size_t ws_size,
                              hipStream_t stream) {
  const float* x        = (const float*)d_in[0];
  const float* x_prev   = (const float*)d_in[1];
  const float* w_off    = (const float*)d_in[2];
  const float* b_off    = (const float*)d_in[3];
  const float* w_def    = (const float*)d_in[4];
  const float* w_cross  = (const float*)d_in[5];
  const float* w_g1     = (const float*)d_in[6];
  const float* b_g1     = (const float*)d_in[7];
  const float* g1_gamma = (const float*)d_in[8];
  const float* g1_beta  = (const float*)d_in[9];
  const float* g1_mean  = (const float*)d_in[10];
  const float* g1_var   = (const float*)d_in[11];
  const float* w_g2     = (const float*)d_in[12];
  const float* b_g2     = (const float*)d_in[13];
  const float* w_out    = (const float*)d_in[14];
  const float* b_out    = (const float*)d_in[15];
  const float* o_gamma  = (const float*)d_in[16];
  const float* o_beta   = (const float*)d_in[17];
  const float* o_mean   = (const float*)d_in[18];
  const float* o_var    = (const float*)d_in[19];
  float* out = (float*)d_out;

  // ---- workspace carve-up (~55 MB) ----
  char* p = (char*)d_ws;
  float* off_buf  = (float*)p;                   p += (size_t)B_ * 18 * HW * 4;
  float* xdense_f = (float*)p;                   p += (size_t)BHW * 256 * 4;
  unsigned short* xbf       = (unsigned short*)p; p += (size_t)BHW * 256 * 2;
  unsigned short* xprevbf   = (unsigned short*)p; p += (size_t)BHW * 256 * 2;
  unsigned short* xdir_bf   = (unsigned short*)p; p += (size_t)BHW * 256 * 2;
  unsigned short* xdense_bf = (unsigned short*)p; p += (size_t)BHW * 256 * 2;
  unsigned short* abuf_bf   = (unsigned short*)p; p += (size_t)BHW * 64 * 2;
  unsigned short* woff_bf   = (unsigned short*)p; p += (size_t)32 * 9 * 256 * 2;
  unsigned short* wg1_bf    = (unsigned short*)p; p += (size_t)64 * 9 * 256 * 2;
  unsigned short* wdef_bf   = (unsigned short*)p; p += (size_t)256 * 9 * 64 * 2;
  unsigned short* wcross_bf = (unsigned short*)p; p += (size_t)256 * 512 * 2;
  unsigned short* wg2_bf    = (unsigned short*)p; p += (size_t)256 * 64 * 2;
  unsigned short* wout_bf   = (unsigned short*)p; p += (size_t)256 * 256 * 2;

  // ---- prep ----
  {
    const int n = BHW * 256;
    k_nchw2nhwc<<<(n + 255) / 256, 256, 0, stream>>>(x,      xbf,     256, n);
    k_nchw2nhwc<<<(n + 255) / 256, 256, 0, stream>>>(x_prev, xprevbf, 256, n);
  }
  { const int n = 32 * 9 * 256;
    k_w3x3_reorder<<<(n + 255) / 256, 256, 0, stream>>>(w_off, woff_bf, 18, 256, n); }
  { const int n = 64 * 9 * 256;
    k_w3x3_reorder<<<(n + 255) / 256, 256, 0, stream>>>(w_g1, wg1_bf, 64, 256, n); }
  { const int n = 256 * 9 * 64;
    k_w3x3_reorder<<<(n + 255) / 256, 256, 0, stream>>>(w_def, wdef_bf, 256, 64, n); }
  { const int n = 256 * 512;
    k_cvt_bf16<<<(n + 255) / 256, 256, 0, stream>>>(w_cross, wcross_bf, n); }
  { const int n = 256 * 64;
    k_cvt_bf16<<<(n + 255) / 256, 256, 0, stream>>>(w_g2, wg2_bf, n); }
  { const int n = 256 * 256;
    k_cvt_bf16<<<(n + 255) / 256, 256, 0, stream>>>(w_out, wout_bf, n); }

  // 1) offset conv (Cout=18, MT=2)
  k_conv3x3<18, 2, false><<<256, 128, 0, stream>>>(
      xbf, woff_bf, b_off, nullptr, nullptr, nullptr, nullptr, off_buf, nullptr);

  // 2) deformable grouped conv
  k_deform<<<1024, 128, 0, stream>>>(xbf, off_buf, wdef_bf, xdir_bf);

  // 3) cross 1x1 (K=512)
  k_dense<<<1024, 128, 0, stream>>>(xdir_bf, xprevbf, wcross_bf, xdense_f, xdense_bf);

  // 4) g1 3x3 conv + BN + SiLU (Cout=64, MT=4)
  k_conv3x3<64, 4, true><<<256, 128, 0, stream>>>(
      xdense_bf, wg1_bf, b_g1, g1_gamma, g1_beta, g1_mean, g1_var, nullptr, abuf_bf);

  // 5) fused gate + out 1x1 + BN + SiLU + residual
  k_attn_out<<<1024, 256, 0, stream>>>(
      abuf_bf, wg2_bf, b_g2, xdense_f, wout_bf, b_out,
      o_gamma, o_beta, o_mean, o_var, x, out);
}
